// SpiralConvBlock_45518063403510
// MI455X (gfx1250) — matrix operand order repaired and checked by reading, not compiled, not run on hardware
//
#include <hip/hip_runtime.h>

// ---------------------------------------------------------------------------
// SpiralConvBlock for MI455X (gfx1250, wave32).
//   block1: LN -> first-order complex IIR scan (== the FFT conv) -> residual
//   block2: LN -> FFN GEMMs on the f16 WMMA pipe (hi/lo split, f32-faithful)
// GEMM tiles are moved by the Tensor Data Mover (tensor_load_to_lds, TENSORcnt)
// with double-buffered LDS stages so DMA overlaps v_wmma_f32_16x16x32_f16.
// ---------------------------------------------------------------------------

typedef _Float16 v16h __attribute__((ext_vector_type(16)));
typedef float    v8f  __attribute__((ext_vector_type(8)));
typedef unsigned int v4u __attribute__((ext_vector_type(4)));
typedef int      v4i  __attribute__((ext_vector_type(4)));
typedef int      v8i  __attribute__((ext_vector_type(8)));

static constexpr int Lseq = 4096;
static constexpr int Bb   = 4;
static constexpr int Dd   = 1024;
static constexpr int FFf  = 4096;
static constexpr int Mm   = Lseq * Bb;      // 16384 rows
static constexpr int CH   = Bb * Dd;        // 4096 scan channels
static constexpr int NCH  = 64;             // chunks
static constexpr int TT   = Lseq / NCH;     // 64 steps per chunk

// ---------------------------------------------------------------------------
// LayerNorm: one wave per row of 1024. SPLIT=false -> f32, SPLIT=true -> f16
// hi/lo pair feeding the WMMA GEMM.
// ---------------------------------------------------------------------------
struct alignas(8) H4 { _Float16 v[4]; };

template <bool SPLIT>
__global__ __launch_bounds__(256) void ln_kernel(
    const float* __restrict__ in, const float* __restrict__ w,
    const float* __restrict__ b, float* __restrict__ outF,
    _Float16* __restrict__ outHi, _Float16* __restrict__ outLo)
{
  const int wid  = threadIdx.x >> 5;
  const int lane = threadIdx.x & 31;
  const int row  = blockIdx.x * 8 + wid;
  const float* p = in + (size_t)row * Dd;

  float4 v[8];
  float s = 0.f, ss = 0.f;
#pragma unroll
  for (int i = 0; i < 8; ++i) {
    v[i] = *(const float4*)(p + i * 128 + lane * 4);
    s  += v[i].x + v[i].y + v[i].z + v[i].w;
    ss += v[i].x * v[i].x + v[i].y * v[i].y + v[i].z * v[i].z + v[i].w * v[i].w;
  }
#pragma unroll
  for (int m = 16; m >= 1; m >>= 1) {
    s  += __shfl_xor(s, m, 32);
    ss += __shfl_xor(ss, m, 32);
  }
  const float mu  = s * (1.f / Dd);
  const float var = ss * (1.f / Dd) - mu * mu;
  const float rs  = rsqrtf(var + 1e-5f);

#pragma unroll
  for (int i = 0; i < 8; ++i) {
    const int d = i * 128 + lane * 4;
    const float4 wv = *(const float4*)(w + d);
    const float4 bv = *(const float4*)(b + d);
    float4 o;
    o.x = (v[i].x - mu) * rs * wv.x + bv.x;
    o.y = (v[i].y - mu) * rs * wv.y + bv.y;
    o.z = (v[i].z - mu) * rs * wv.z + bv.z;
    o.w = (v[i].w - mu) * rs * wv.w + bv.w;
    const size_t idx = (size_t)row * Dd + d;
    if (SPLIT) {
      H4 hi, lo;
      const float of[4] = {o.x, o.y, o.z, o.w};
#pragma unroll
      for (int j = 0; j < 4; ++j) {
        _Float16 h = (_Float16)of[j];
        hi.v[j] = h;
        lo.v[j] = (_Float16)(of[j] - (float)h);
      }
      *(H4*)(outHi + idx) = hi;
      *(H4*)(outLo + idx) = lo;
    } else {
      *(float4*)(outF + idx) = o;
    }
  }
}

// ---------------------------------------------------------------------------
// Spiral-conv as a chunked linear scan:  s[t] = c*s[t-1] + h[t], s[-1]=lc
// ---------------------------------------------------------------------------
__device__ inline void chan_coef(const float* lmlg, const float* theta, int d,
                                 float& cr, float& ci) {
  const float g = expf(-expf(lmlg[d]));
  cr = g * cosf(theta[d]);
  ci = g * sinf(theta[d]);
}

__global__ __launch_bounds__(256) void scan_local_kernel(
    const float* __restrict__ h, const float* __restrict__ lmlg,
    const float* __restrict__ theta, float2* __restrict__ S)
{
  const int ch = blockIdx.y * 256 + threadIdx.x;   // b*D + d
  const int chunk = blockIdx.x;
  float cr, ci;
  chan_coef(lmlg, theta, ch & (Dd - 1), cr, ci);
  float sr = 0.f, si = 0.f;
  const float* p = h + (size_t)chunk * TT * CH + ch;
#pragma unroll 4
  for (int i = 0; i < TT; ++i) {
    const float hv = p[(size_t)i * CH];
    const float nr = fmaf(cr, sr, fmaf(-ci, si, hv));
    const float ni = fmaf(cr, si, ci * sr);
    sr = nr; si = ni;
  }
  S[(size_t)chunk * CH + ch] = make_float2(sr, si);
}

__global__ __launch_bounds__(256) void scan_prefix_kernel(
    const float* __restrict__ lmlg, const float* __restrict__ theta,
    const float* __restrict__ lci, const float2* __restrict__ S,
    float2* __restrict__ Inc)
{
  const int ch = blockIdx.x * 256 + threadIdx.x;
  const int d  = ch & (Dd - 1);
  float cr, ci;
  chan_coef(lmlg, theta, d, cr, ci);
  float tr = cr, ti = ci;                 // c^64 by 6 squarings
#pragma unroll
  for (int i = 0; i < 6; ++i) {
    const float nr = tr * tr - ti * ti;
    const float ni = 2.f * tr * ti;
    tr = nr; ti = ni;
  }
  float ir = lci[d], ii = 0.f;            // s[-1] = last_conv (real)
  for (int k = 0; k < NCH; ++k) {
    Inc[(size_t)k * CH + ch] = make_float2(ir, ii);
    const float2 sk = S[(size_t)k * CH + ch];
    const float nr = tr * ir - ti * ii + sk.x;
    const float ni = tr * ii + ti * ir + sk.y;
    ir = nr; ii = ni;
  }
}

__global__ __launch_bounds__(256) void scan_apply_kernel(
    const float* __restrict__ h, const float* __restrict__ x,
    const float* __restrict__ lmlg, const float* __restrict__ theta,
    const float2* __restrict__ Inc, float* __restrict__ x1)
{
  const int ch = blockIdx.y * 256 + threadIdx.x;
  const int chunk = blockIdx.x;
  float cr, ci;
  chan_coef(lmlg, theta, ch & (Dd - 1), cr, ci);
  const float2 s0 = Inc[(size_t)chunk * CH + ch];
  float sr = s0.x, si = s0.y;
  const size_t base = (size_t)chunk * TT * CH + ch;
#pragma unroll 4
  for (int i = 0; i < TT; ++i) {
    const size_t idx = base + (size_t)i * CH;
    const float hv = h[idx];
    const float nr = fmaf(cr, sr, fmaf(-ci, si, hv));
    const float ni = fmaf(cr, si, ci * sr);
    sr = nr; si = ni;
    x1[idx] = sr * hv + x[idx];           // Re(state)*h + residual
  }
}

// ---------------------------------------------------------------------------
// Split W[K][N] f32 into transposed f16 hi/lo pair Wt[N][K].
// ---------------------------------------------------------------------------
__global__ __launch_bounds__(256) void split_transpose_kernel(
    const float* __restrict__ W, _Float16* __restrict__ Hi,
    _Float16* __restrict__ Lo, int K, int N)
{
  __shared__ float tile[32][33];
  const int tx = threadIdx.x & 31, ty = threadIdx.x >> 5;
  const int n0 = blockIdx.x * 32, k0 = blockIdx.y * 32;
#pragma unroll
  for (int j = 0; j < 4; ++j) {
    const int k = k0 + ty + j * 8;
    tile[ty + j * 8][tx] = W[(size_t)k * N + n0 + tx];
  }
  __syncthreads();
#pragma unroll
  for (int j = 0; j < 4; ++j) {
    const int n = n0 + ty + j * 8;
    const float v = tile[tx][ty + j * 8];
    const _Float16 hv = (_Float16)v;
    const size_t o = (size_t)n * K + k0 + tx;
    Hi[o] = hv;
    Lo[o] = (_Float16)(v - (float)hv);
  }
}

// ---------------------------------------------------------------------------
// TDM: issue one 128x32 f16 tile load (row stride K elems) into LDS with
// +16B row padding (pad_interval code 3 = 16 dwords, pad_amount code 3 = 4
// dwords) -> matches the LDK=40-half LDS pitch. D# fields per ISA 8.3/8.4.
// ---------------------------------------------------------------------------
static constexpr int BM = 128, BN = 128, BK = 32;
static constexpr int LDK = 40;             // 32 + 8 halfs pad per row

__device__ __forceinline__ void tdm_tile_load(unsigned ldsOff,
                                              const _Float16* gbase, int K)
{
  const unsigned long long ga = (unsigned long long)(uintptr_t)gbase;
  v4u g0;
  g0.x = 1u;                                           // count=1, user desc
  g0.y = ldsOff;                                       // lds_addr
  g0.z = (unsigned)ga;                                 // global_addr[31:0]
  g0.w = (unsigned)((ga >> 32) & 0x01ffffffull)        // global_addr[56:32]
       | 0x80000000u;                                  // type=2 ("image")
  const unsigned tK = (unsigned)K;                     // tensor_dim0
  const unsigned tR = 1u << 24;                        // tensor_dim1 (ample)
  const unsigned d0 = (1u << 16)                       // data_size = 2B
                    | (1u << 20)                       // pad_enable
                    | (3u << 22)                       // pad_interval: 16 dw
                    | (3u << 25);                      // pad_amount: 4 dw
  const unsigned d1 = (tK & 0xffffu) << 16;            // tensor_dim0[15:0]
  const unsigned d2 = (tK >> 16) | ((tR & 0xffffu) << 16);
  const unsigned d3 = (tR >> 16) | ((unsigned)BK << 16); // tile_dim0 = 32
  const unsigned d4 = (unsigned)BM;                    // tile_dim1=128, dim2=0
  const unsigned d5 = tK;                              // dim0_stride[31:0] = K
  const v8i g1 = {(int)d0, (int)d1, (int)d2, (int)d3,
                  (int)d4, (int)d5, 0, 0};
  const v4i z4 = {0, 0, 0, 0};
#if defined(__clang_major__) && __clang_major__ >= 23
  const v8i z8 = {0, 0, 0, 0, 0, 0, 0, 0};
  __builtin_amdgcn_tensor_load_to_lds(g0, g1, z4, z4, z8, 0);
#else
  __builtin_amdgcn_tensor_load_to_lds(g0, g1, z4, z4, 0);
#endif
}

// ---------------------------------------------------------------------------
// WMMA GEMM:  C[M][N] = A[M][K] * B[N][K]^T (+bias, then SiLU or residual)
// f16 split: acc += Ah*Bh + Ah*Bl + Al*Bh  (f32 accumulate, ~f32-faithful)
// Block tile 128x128x32, 8 waves, wave tile 32x64 = 2x4 WMMA tiles.
// Double-buffered LDS stages fed by the Tensor Data Mover; wave 0 issues the
// DMA + s_wait_tensorcnt, workgroup barrier publishes the stage.
// ---------------------------------------------------------------------------
template <int EPI>  // 0: bias+SiLU -> hi/lo halves   1: bias+residual -> f32
__global__ __launch_bounds__(256) void gemm_wmma_kernel(
    const _Float16* __restrict__ AHi, const _Float16* __restrict__ ALo,
    const _Float16* __restrict__ BHi, const _Float16* __restrict__ BLo,
    const float* __restrict__ bias, const float* __restrict__ resid,
    _Float16* __restrict__ OHi, _Float16* __restrict__ OLo,
    float* __restrict__ OF, int M, int N, int K)
{
  __shared__ _Float16 sA[2][2][BM][LDK];   // [stage][hi/lo][row][k]
  __shared__ _Float16 sB[2][2][BN][LDK];

  const int tid = threadIdx.x;
  const int lane = tid & 31, w = tid >> 5;
  const int wr = w & 3, wc = w >> 2;
  const int blockM = blockIdx.y * BM, blockN = blockIdx.x * BN;
  const int lr = lane & 15, hf = lane >> 4;
  const int m0 = wr * 32, n0 = wc * 64;

  const size_t aBase = (size_t)blockM * K;
  const size_t bBase = (size_t)blockN * K;
  const unsigned ldsA = (unsigned)(uintptr_t)&sA[0][0][0][0];
  const unsigned ldsB = (unsigned)(uintptr_t)&sB[0][0][0][0];
  const unsigned STG = 2u * BM * LDK * 2u;   // bytes per stage (hi+lo)
  const unsigned ARR = (unsigned)BM * LDK * 2u;

  const v8f vzero = {0.f, 0.f, 0.f, 0.f, 0.f, 0.f, 0.f, 0.f};
  v8f acc[2][4];
#pragma unroll
  for (int mt = 0; mt < 2; ++mt)
#pragma unroll
    for (int nt = 0; nt < 4; ++nt) acc[mt][nt] = vzero;

  const int nk = K / BK;
  if (w == 0) {                               // prologue: stage 0 <- kb 0
    tdm_tile_load(ldsA,             AHi + aBase, K);
    tdm_tile_load(ldsA + ARR,       ALo + aBase, K);
    tdm_tile_load(ldsB,             BHi + bBase, K);
    tdm_tile_load(ldsB + ARR,       BLo + bBase, K);
  }

  for (int kbi = 0; kbi < nk; ++kbi) {
    const int cur = kbi & 1;
    if (w == 0) __builtin_amdgcn_s_wait_tensorcnt((short)0);
    __syncthreads();                          // stage `cur` ready for all waves
    if (w == 0 && kbi + 1 < nk) {             // prefetch stage cur^1 <- kb+BK
      const size_t ko = (size_t)(kbi + 1) * BK;
      const unsigned so = (unsigned)(cur ^ 1) * STG;
      tdm_tile_load(ldsA + so,       AHi + aBase + ko, K);
      tdm_tile_load(ldsA + so + ARR, ALo + aBase + ko, K);
      tdm_tile_load(ldsB + so,       BHi + bBase + ko, K);
      tdm_tile_load(ldsB + so + ARR, BLo + bBase + ko, K);
    }

    // --- fragments (ISA 7.12.2 16-bit layouts)
    v16h aF[2][2];
#pragma unroll
    for (int mt = 0; mt < 2; ++mt)
#pragma unroll
      for (int arr = 0; arr < 2; ++arr) {
        const _Float16* p = &sA[cur][arr][m0 + mt * 16 + lr][hf * 8];
        v16h t;
        *(uint4*)&t         = *(const uint4*)p;        // K = hf*8 .. +7
        *(((uint4*)&t) + 1) = *(const uint4*)(p + 16); // K = 16+hf*8 .. +7
        aF[mt][arr] = t;
      }
    v16h bF[4][2];
#pragma unroll
    for (int nt = 0; nt < 4; ++nt)
#pragma unroll
      for (int arr = 0; arr < 2; ++arr) {
        const _Float16* p = &sB[cur][arr][n0 + nt * 16 + lr][hf * 8];
        v16h t;
        *(uint4*)&t         = *(const uint4*)p;        // K = hf*8 .. +7
        *(((uint4*)&t) + 1) = *(const uint4*)(p + 16); // K = 16+hf*8 .. +7
        bF[nt][arr] = t;
      }

#pragma unroll
    for (int mt = 0; mt < 2; ++mt)
#pragma unroll
      for (int nt = 0; nt < 4; ++nt) {
        v8f c = acc[mt][nt];
        c = __builtin_amdgcn_wmma_f32_16x16x32_f16(false, aF[mt][0], false, bF[nt][0], (short)0, c, false, false);
        c = __builtin_amdgcn_wmma_f32_16x16x32_f16(false, aF[mt][0], false, bF[nt][1], (short)0, c, false, false);
        c = __builtin_amdgcn_wmma_f32_16x16x32_f16(false, aF[mt][1], false, bF[nt][0], (short)0, c, false, false);
        acc[mt][nt] = c;
      }
    __syncthreads();                          // done reading stage `cur`
  }

  // --- epilogue (C/D layout: VGPR r -> row r + 8*hf, col = lane&15)
#pragma unroll
  for (int mt = 0; mt < 2; ++mt)
#pragma unroll
    for (int nt = 0; nt < 4; ++nt) {
      const int col = blockN + n0 + nt * 16 + lr;
      const float bv = bias[col];
#pragma unroll
      for (int r = 0; r < 8; ++r) {
        const int row = blockM + m0 + mt * 16 + r + hf * 8;
        float v = acc[mt][nt][r] + bv;
        const size_t o = (size_t)row * N + col;
        if (EPI == 0) {
          const float s = v / (1.f + __expf(-v));   // SiLU
          const _Float16 hi = (_Float16)s;
          OHi[o] = hi;
          OLo[o] = (_Float16)(s - (float)hi);
        } else {
          OF[o] = v + resid[o];
        }
      }
    }
}

// ---------------------------------------------------------------------------
extern "C" void kernel_launch(void* const* d_in, const int* in_sizes, int n_in,
                              void* d_out, int out_size, void* d_ws, size_t ws_size,
                              hipStream_t stream) {
  (void)in_sizes; (void)n_in; (void)out_size; (void)ws_size;
  const float* x     = (const float*)d_in[0];
  const float* lmlg  = (const float*)d_in[1];
  const float* theta = (const float*)d_in[2];
  const float* lci   = (const float*)d_in[3];
  const float* lnw   = (const float*)d_in[4];
  const float* lnb   = (const float*)d_in[5];
  const float* w1    = (const float*)d_in[6];
  const float* b1    = (const float*)d_in[7];
  const float* w2    = (const float*)d_in[8];
  const float* b2    = (const float*)d_in[9];

  // workspace layout (~420 MB)
  char* p = (char*)d_ws;
  float*   x1    = (float*)p;  p += (size_t)Mm * Dd * 4;          // 64 MB
  float2*  S     = (float2*)p; p += (size_t)NCH * CH * 8;         //  2 MB
  float2*  Inc   = (float2*)p; p += (size_t)NCH * CH * 8;         //  2 MB
  _Float16* w1tHi = (_Float16*)p; p += (size_t)FFf * Dd * 2;      //  8 MB
  _Float16* w1tLo = (_Float16*)p; p += (size_t)FFf * Dd * 2;
  _Float16* w2tHi = (_Float16*)p; p += (size_t)Dd * FFf * 2;
  _Float16* w2tLo = (_Float16*)p; p += (size_t)Dd * FFf * 2;
  float*   hF    = (float*)p;  p += (size_t)Mm * Dd * 4;          // 64 MB (reused)
  _Float16* hHi  = (_Float16*)hF;          // reuse hF region after scan
  _Float16* hLo  = hHi + (size_t)Mm * Dd;
  _Float16* actHi = (_Float16*)p; p += (size_t)Mm * FFf * 2;      // 128 MB
  _Float16* actLo = (_Float16*)p; p += (size_t)Mm * FFf * 2;      // 128 MB

  // weights -> transposed f16 hi/lo
  split_transpose_kernel<<<dim3(FFf / 32, Dd / 32), 256, 0, stream>>>(w1, w1tHi, w1tLo, Dd, FFf);
  split_transpose_kernel<<<dim3(Dd / 32, FFf / 32), 256, 0, stream>>>(w2, w2tHi, w2tLo, FFf, Dd);

  // block 1: LN -> scan -> residual
  ln_kernel<false><<<Mm / 8, 256, 0, stream>>>(x, lnw, lnb, hF, nullptr, nullptr);
  scan_local_kernel<<<dim3(NCH, CH / 256), 256, 0, stream>>>(hF, lmlg, theta, S);
  scan_prefix_kernel<<<CH / 256, 256, 0, stream>>>(lmlg, theta, lci, S, Inc);
  scan_apply_kernel<<<dim3(NCH, CH / 256), 256, 0, stream>>>(hF, x, lmlg, theta, Inc, x1);

  // block 2: LN -> FFN(SiLU) -> residual
  ln_kernel<true><<<Mm / 8, 256, 0, stream>>>(x1, lnw, lnb, nullptr, hHi, hLo);
  gemm_wmma_kernel<0><<<dim3(FFf / BN, Mm / BM), 256, 0, stream>>>(
      hHi, hLo, w1tHi, w1tLo, b1, nullptr, actHi, actLo, nullptr, Mm, FFf, Dd);
  gemm_wmma_kernel<1><<<dim3(Dd / BN, Mm / BM), 256, 0, stream>>>(
      actHi, actLo, w2tHi, w2tLo, b2, x1, nullptr, nullptr, (float*)d_out, Mm, Dd, FFf);
}
